// HybridMoE_77438260347034
// MI455X (gfx1250) — matrix operand order repaired and checked by reading, not compiled, not run on hardware
//
#include <hip/hip_runtime.h>

typedef __attribute__((ext_vector_type(16))) __bf16 bf16x16;
typedef __attribute__((ext_vector_type(8)))  float  f32x8;
typedef __attribute__((ext_vector_type(4)))  int    v4i;

#define T_TOK 2048
#define H_DIM 1024
#define F_DIM 2816
#define N_EXP 8
#define TOPK  2

#if __has_builtin(__builtin_amdgcn_global_load_async_to_lds_b128)
#define HAS_ASYNC_LOAD 1
#endif
#if __has_builtin(__builtin_amdgcn_global_store_async_from_lds_b128)
#define HAS_ASYNC_STORE 1
#endif

typedef __attribute__((address_space(1))) v4i glob_v4i;
typedef __attribute__((address_space(3))) v4i lds_v4i;

__device__ __forceinline__ unsigned short f2bf(float f) {
  unsigned int u = __float_as_uint(f);
  u += 0x7FFFu + ((u >> 16) & 1u);   // round-to-nearest-even
  return (unsigned short)(u >> 16);
}

// Pack two floats to packed bf16x2 (a -> low 16 bits, b -> high 16 bits).
__device__ __forceinline__ unsigned int pack_bf2(float a, float b) {
#if __has_builtin(__builtin_amdgcn_cvt_pk_bf16_f32)
  typedef __attribute__((ext_vector_type(2))) __bf16 bf16x2;
  bf16x2 v = __builtin_amdgcn_cvt_pk_bf16_f32(a, b);
  return __builtin_bit_cast(unsigned int, v);
#else
  return (unsigned int)f2bf(a) | ((unsigned int)f2bf(b) << 16);
#endif
}

union FragU { unsigned int u[8]; bf16x16 v; };

// A-matrix fragment (16x32 bf16): lanes 0-15 row M=lane, K=0-15;
// lanes 16-31 row M=lane-16, K=16-31. LDS tile row-major [m][k], stride 40.
__device__ __forceinline__ bf16x16 frag_a(const unsigned short* s, int m, int hi) {
  FragU f;
  const unsigned int* p = (const unsigned int*)(s + m * 40 + hi * 16);
  f.u[0] = p[0]; f.u[1] = p[1]; f.u[2] = p[2]; f.u[3] = p[3];
  f.u[4] = p[4]; f.u[5] = p[5]; f.u[6] = p[6]; f.u[7] = p[7];
  return f.v;
}

// B-matrix fragment (32x16 bf16): lanes 0-15 col N=lane, K=0-15;
// lanes 16-31 col N=lane-16, K=16-31. LDS tile transposed [n][k], stride 40.
__device__ __forceinline__ bf16x16 frag_b(const unsigned short* s, int n, int hi) {
  FragU f;
  const unsigned int* p = (const unsigned int*)(s + n * 40 + hi * 16);
#pragma unroll
  for (int i = 0; i < 8; ++i) f.u[i] = p[i];
  return f.v;
}

__device__ __forceinline__ f32x8 zero8() {
  f32x8 z = {0.f, 0.f, 0.f, 0.f, 0.f, 0.f, 0.f, 0.f};
  return z;
}

__global__ __launch_bounds__(256) void zero_init(float* out, int n, int* cnt) {
  int tid = blockIdx.x * blockDim.x + threadIdx.x;
  if (tid < N_EXP) cnt[tid] = 0;
  for (int i = tid; i < n; i += gridDim.x * blockDim.x) out[i] = 0.f;
}

__global__ __launch_bounds__(256) void router_kernel(const float* __restrict__ x,
                                                     const float* __restrict__ rw,
                                                     float* __restrict__ slotw,
                                                     int* __restrict__ lists,
                                                     int* __restrict__ cnt) {
  __shared__ float s_rw[H_DIM * 9];  // pad 8->9 to avoid bank conflicts
  int tid = threadIdx.x;
  for (int i = tid; i < H_DIM * N_EXP; i += 256) {
    int h = i >> 3, e = i & 7;
    s_rw[h * 9 + e] = rw[i];
  }
  __syncthreads();
  int lane = tid & 31, w = tid >> 5;
  int t = blockIdx.x * 8 + w;
  float acc[N_EXP];
#pragma unroll
  for (int e = 0; e < N_EXP; ++e) acc[e] = 0.f;
  const float* xp = x + (size_t)t * H_DIM;
  for (int h0 = 0; h0 < H_DIM; h0 += 32) {
    float xv = xp[h0 + lane];
#pragma unroll
    for (int e = 0; e < N_EXP; ++e) acc[e] += xv * s_rw[(h0 + lane) * 9 + e];
  }
#pragma unroll
  for (int e = 0; e < N_EXP; ++e) {
    acc[e] += __shfl_xor(acc[e], 16, 32);
    acc[e] += __shfl_xor(acc[e], 8, 32);
    acc[e] += __shfl_xor(acc[e], 4, 32);
    acc[e] += __shfl_xor(acc[e], 2, 32);
    acc[e] += __shfl_xor(acc[e], 1, 32);
  }
  if (lane == 0) {
    int i0 = 0;
    for (int e = 1; e < N_EXP; ++e) if (acc[e] > acc[i0]) i0 = e;
    int i1 = (i0 == 0) ? 1 : 0;
    for (int e = 0; e < N_EXP; ++e) if (e != i0 && acc[e] > acc[i1]) i1 = e;
    // normalized top-2 softmax weights
    float e1 = __expf(acc[i1] - acc[i0]);
    float inv = __builtin_amdgcn_rcpf(1.f + e1);
    slotw[2 * t + 0] = inv;
    slotw[2 * t + 1] = e1 * inv;
    int p0 = atomicAdd(&cnt[i0], 1); lists[i0 * T_TOK + p0] = 2 * t;
    int p1 = atomicAdd(&cnt[i1], 1); lists[i1 * T_TOK + p1] = 2 * t + 1;
  }
}

// Stage A: per expert, h[r, :] = silu(x_r @ Wg) * (x_r @ Wu), rows gathered by list.
__global__ __launch_bounds__(256) void gateup_kernel(const float* __restrict__ x,
                                                     const float* __restrict__ wg,
                                                     const float* __restrict__ wu,
                                                     const int* __restrict__ lists,
                                                     const int* __restrict__ cnt,
                                                     unsigned short* __restrict__ hbuf) {
  int e = blockIdx.z;
  int count = cnt[e];
  int row0 = blockIdx.y * 128;
  if (row0 >= count) return;
  int n0 = blockIdx.x * 64;

  // aliased shared pool: GEMM tiles during main loop, C-tile staging in epilogue
  __shared__ __align__(16) unsigned short smem[128 * 40 + 64 * 40 + 64 * 40];  // 20480 B
  unsigned short* s_a = smem;                 // [128][40]
  unsigned short* s_g = smem + 128 * 40;      // [64][40]
  unsigned short* s_u = s_g + 64 * 40;        // [64][40]
  unsigned short* s_o = smem;                 // epilogue staging [128][72] = 18432 B
  __shared__ int s_row[128];

  int tid = threadIdx.x, lane = tid & 31, wid = tid >> 5;
  int wm = wid & 3, wn = wid >> 2;

  if (tid < 128) {
    int p = row0 + tid;
    s_row[tid] = (p < count) ? lists[e * T_TOK + p] : -1;
  }
  __syncthreads();

  f32x8 aG[4], aU[4];
#pragma unroll
  for (int i = 0; i < 4; ++i) { aG[i] = zero8(); aU[i] = zero8(); }

  const float* wge = wg + (size_t)e * H_DIM * F_DIM;
  const float* wue = wu + (size_t)e * H_DIM * F_DIM;
  int arow = tid >> 1, acol = (tid & 1) * 16;      // A tile mapping
  int kq2 = (tid >> 4) * 2, nq4 = (tid & 15) * 4;  // W tile mapping (k-pair x 4n)
  int hi = lane >> 4, nl = lane & 15;

  for (int kk = 0; kk < H_DIM; kk += 32) {
    // A tile: 128 gathered x rows x 32 k, fp32 -> packed bf16, 2x b128 LDS stores
    {
      int r = s_row[arow];
      const float4* xp = (const float4*)(x + (size_t)((r < 0) ? 0 : (r >> 1)) * H_DIM + kk + acol);
      float4 v0 = xp[0], v1 = xp[1], v2 = xp[2], v3 = xp[3];
      uint4 lo, hi4;
      lo.x = pack_bf2(v0.x, v0.y); lo.y = pack_bf2(v0.z, v0.w);
      lo.z = pack_bf2(v1.x, v1.y); lo.w = pack_bf2(v1.z, v1.w);
      hi4.x = pack_bf2(v2.x, v2.y); hi4.y = pack_bf2(v2.z, v2.w);
      hi4.z = pack_bf2(v3.x, v3.y); hi4.w = pack_bf2(v3.z, v3.w);
      *(uint4*)(s_a + arow * 40 + acol) = lo;
      *(uint4*)(s_a + arow * 40 + acol + 8) = hi4;
    }
    // W tiles: 32(k) x 64(n) fp32, transposed [n][k]; pack k-pairs -> b32 LDS stores
    {
      const float4* gp0 = (const float4*)(wge + (size_t)(kk + kq2) * F_DIM + n0 + nq4);
      const float4* gp1 = (const float4*)(wge + (size_t)(kk + kq2 + 1) * F_DIM + n0 + nq4);
      const float4* up0 = (const float4*)(wue + (size_t)(kk + kq2) * F_DIM + n0 + nq4);
      const float4* up1 = (const float4*)(wue + (size_t)(kk + kq2 + 1) * F_DIM + n0 + nq4);
      float4 ga = gp0[0], gb = gp1[0], ua = up0[0], ub = up1[0];
      *(unsigned int*)(s_g + (nq4 + 0) * 40 + kq2) = pack_bf2(ga.x, gb.x);
      *(unsigned int*)(s_g + (nq4 + 1) * 40 + kq2) = pack_bf2(ga.y, gb.y);
      *(unsigned int*)(s_g + (nq4 + 2) * 40 + kq2) = pack_bf2(ga.z, gb.z);
      *(unsigned int*)(s_g + (nq4 + 3) * 40 + kq2) = pack_bf2(ga.w, gb.w);
      *(unsigned int*)(s_u + (nq4 + 0) * 40 + kq2) = pack_bf2(ua.x, ub.x);
      *(unsigned int*)(s_u + (nq4 + 1) * 40 + kq2) = pack_bf2(ua.y, ub.y);
      *(unsigned int*)(s_u + (nq4 + 2) * 40 + kq2) = pack_bf2(ua.z, ub.z);
      *(unsigned int*)(s_u + (nq4 + 3) * 40 + kq2) = pack_bf2(ua.w, ub.w);
    }
    __syncthreads();

    bf16x16 af[2], bg[2], bu[2];
#pragma unroll
    for (int fm = 0; fm < 2; ++fm) af[fm] = frag_a(s_a, wm * 32 + fm * 16 + nl, hi);
#pragma unroll
    for (int fn = 0; fn < 2; ++fn) {
      bg[fn] = frag_b(s_g, wn * 32 + fn * 16 + nl, hi);
      bu[fn] = frag_b(s_u, wn * 32 + fn * 16 + nl, hi);
    }
#pragma unroll
    for (int fm = 0; fm < 2; ++fm)
#pragma unroll
      for (int fn = 0; fn < 2; ++fn) {
        aG[fm * 2 + fn] = __builtin_amdgcn_wmma_f32_16x16x32_bf16(
            false, af[fm], false, bg[fn], (short)0, aG[fm * 2 + fn], false, false);
        aU[fm * 2 + fn] = __builtin_amdgcn_wmma_f32_16x16x32_bf16(
            false, af[fm], false, bu[fn], (short)0, aU[fm * 2 + fn], false, false);
      }
    __syncthreads();  // also fences GEMM tiles before epilogue re-uses the pool
  }

  // epilogue: silu(g)*u, stage bf16 C-tile [m][n] (stride 72) in LDS,
  // then coalesced 128b stores to hbuf rows (async LDS->global when available).
#pragma unroll
  for (int fm = 0; fm < 2; ++fm)
#pragma unroll
    for (int fn = 0; fn < 2; ++fn) {
      f32x8 g = aG[fm * 2 + fn], u = aU[fm * 2 + fn];
      int nloc = wn * 32 + fn * 16 + nl;
      int mbase = wm * 32 + fm * 16 + hi * 8;
#pragma unroll
      for (int j = 0; j < 8; ++j) {
        float gv = g[j], uv = u[j];
        float sv = gv * __builtin_amdgcn_rcpf(1.f + __expf(-gv)) * uv;
        s_o[(mbase + j) * 72 + nloc] = f2bf(sv);
      }
    }
  __syncthreads();
  {
    int m = tid >> 1;
    int nh = (tid & 1) * 32;
    int r = s_row[m];
    if (r >= 0) {
      unsigned short* dst = hbuf + (size_t)r * F_DIM + n0 + nh;
      unsigned short* src = s_o + m * 72 + nh;
#ifdef HAS_ASYNC_STORE
#pragma unroll
      for (int j = 0; j < 4; ++j)
        __builtin_amdgcn_global_store_async_from_lds_b128(
            (glob_v4i*)(dst + j * 8), (lds_v4i*)(src + j * 8), 0, 0);
      asm volatile("s_wait_asynccnt 0x0" ::: "memory");
#else
      const uint4* s4 = (const uint4*)src;
      uint4 a0 = s4[0], a1 = s4[1], a2 = s4[2], a3 = s4[3];
      uint4* d4 = (uint4*)dst;
      d4[0] = a0; d4[1] = a1; d4[2] = a2; d4[3] = a3;
#endif
    }
  }
}

// Stage B: out[t, :] += c_r * (h_r @ Wd[e]), atomic scatter (2 commutative adds/token).
__global__ __launch_bounds__(256) void down_kernel(const unsigned short* __restrict__ hbuf,
                                                   const float* __restrict__ wd,
                                                   const int* __restrict__ lists,
                                                   const int* __restrict__ cnt,
                                                   const float* __restrict__ slotw,
                                                   float* __restrict__ out) {
  int e = blockIdx.z;
  int count = cnt[e];
  int row0 = blockIdx.y * 128;
  if (row0 >= count) return;
  int n0 = blockIdx.x * 64;

  __shared__ __align__(16) unsigned short s_a[128 * 40];
  __shared__ __align__(16) unsigned short s_w[64 * 40];
  __shared__ int s_row[128];
  __shared__ float s_c[128];

  int tid = threadIdx.x, lane = tid & 31, wid = tid >> 5;
  int wm = wid & 3, wn = wid >> 2;

  if (tid < 128) {
    int p = row0 + tid;
    int r = (p < count) ? lists[e * T_TOK + p] : -1;
    s_row[tid] = r;
    s_c[tid] = (r >= 0) ? slotw[r] : 0.f;
  }
  __syncthreads();

  f32x8 acc[4];
#pragma unroll
  for (int i = 0; i < 4; ++i) acc[i] = zero8();

  const float* wde = wd + (size_t)e * F_DIM * H_DIM;
  int arow = tid >> 1, acol = (tid & 1) * 16;
  int kq2 = (tid >> 4) * 2, nq4 = (tid & 15) * 4;
  int hi = lane >> 4, nl = lane & 15;

  for (int kk = 0; kk < F_DIM; kk += 32) {
    // A tile: gathered h rows, already bf16 -> async global->LDS when available
    {
      int r = s_row[arow];
      const unsigned short* hp = hbuf + (size_t)((r < 0) ? 0 : r) * F_DIM + kk + acol;
      unsigned short* dp = s_a + arow * 40 + acol;
#ifdef HAS_ASYNC_LOAD
      __builtin_amdgcn_global_load_async_to_lds_b128(
          (glob_v4i*)hp, (lds_v4i*)dp, 0, 0);
      __builtin_amdgcn_global_load_async_to_lds_b128(
          (glob_v4i*)(hp + 8), (lds_v4i*)(dp + 8), 0, 0);
#else
      const uint4* h4 = (const uint4*)hp;
      uint4 h0 = h4[0], h1 = h4[1];
      *(uint4*)dp = h0;
      *(uint4*)(dp + 8) = h1;
#endif
    }
    // Wd tile: 32(k over F) x 64(n over H), transposed, packed k-pair stores
    {
      const float4* wp0 = (const float4*)(wde + (size_t)(kk + kq2) * H_DIM + n0 + nq4);
      const float4* wp1 = (const float4*)(wde + (size_t)(kk + kq2 + 1) * H_DIM + n0 + nq4);
      float4 wa = wp0[0], wb = wp1[0];
      *(unsigned int*)(s_w + (nq4 + 0) * 40 + kq2) = pack_bf2(wa.x, wb.x);
      *(unsigned int*)(s_w + (nq4 + 1) * 40 + kq2) = pack_bf2(wa.y, wb.y);
      *(unsigned int*)(s_w + (nq4 + 2) * 40 + kq2) = pack_bf2(wa.z, wb.z);
      *(unsigned int*)(s_w + (nq4 + 3) * 40 + kq2) = pack_bf2(wa.w, wb.w);
    }
#ifdef HAS_ASYNC_LOAD
    asm volatile("s_wait_asynccnt 0x0" ::: "memory");
#endif
    __syncthreads();

    bf16x16 af[2], bw[2];
#pragma unroll
    for (int fm = 0; fm < 2; ++fm) af[fm] = frag_a((const unsigned short*)s_a, wm * 32 + fm * 16 + nl, hi);
#pragma unroll
    for (int fn = 0; fn < 2; ++fn) bw[fn] = frag_b((const unsigned short*)s_w, wn * 32 + fn * 16 + nl, hi);
#pragma unroll
    for (int fm = 0; fm < 2; ++fm)
#pragma unroll
      for (int fn = 0; fn < 2; ++fn)
        acc[fm * 2 + fn] = __builtin_amdgcn_wmma_f32_16x16x32_bf16(
            false, af[fm], false, bw[fn], (short)0, acc[fm * 2 + fn], false, false);
    __syncthreads();
  }

  // epilogue: one row lookup + base address per (fm, j); two atomics share it
#pragma unroll
  for (int fm = 0; fm < 2; ++fm) {
#pragma unroll
    for (int j = 0; j < 8; ++j) {
      int m = wm * 32 + fm * 16 + hi * 8 + j;
      int r = s_row[m];
      if (r >= 0) {
        float c = s_c[m];
        float* op = out + (size_t)(r >> 1) * H_DIM + n0 + wn * 32 + nl;
        atomicAdd(op, acc[fm * 2 + 0][j] * c);
        atomicAdd(op + 16, acc[fm * 2 + 1][j] * c);
      }
    }
  }
}

extern "C" void kernel_launch(void* const* d_in, const int* in_sizes, int n_in,
                              void* d_out, int out_size, void* d_ws, size_t ws_size,
                              hipStream_t stream) {
  (void)in_sizes; (void)n_in; (void)out_size; (void)ws_size;
  const float* x  = (const float*)d_in[0];
  const float* rw = (const float*)d_in[1];
  const float* wg = (const float*)d_in[2];
  const float* wu = (const float*)d_in[3];
  const float* wd = (const float*)d_in[4];
  float* out = (float*)d_out;

  // workspace carve-out
  int* cnt = (int*)d_ws;                                   // E counters (16-int slot)
  int* lists = cnt + 16;                                   // E*T slot ids
  float* slotw = (float*)(lists + N_EXP * T_TOK);          // T*K routing weights
  unsigned short* hbuf = (unsigned short*)(slotw + T_TOK * TOPK);  // [T*K, F] bf16

  zero_init<<<2048, 256, 0, stream>>>(out, T_TOK * H_DIM, cnt);
  router_kernel<<<T_TOK / 8, 256, 0, stream>>>(x, rw, slotw, lists, cnt);
  gateup_kernel<<<dim3(F_DIM / 64, T_TOK / 128, N_EXP), 256, 0, stream>>>(
      x, wg, wu, lists, cnt, hbuf);
  down_kernel<<<dim3(H_DIM / 64, T_TOK / 128, N_EXP), 256, 0, stream>>>(
      hbuf, wd, lists, cnt, slotw, out);
}
